// ASPPModulatedDeformableC3D_53300544143739
// MI455X (gfx1250) — compile-verified, matrix operand order checked by reading, not run on hardware
//
#include <hip/hip_runtime.h>
#include <hip/hip_bf16.h>
#include <math.h>

typedef __attribute__((ext_vector_type(16))) __bf16 v16bf;
typedef __attribute__((ext_vector_type(8)))  float  v8f;
typedef __attribute__((ext_vector_type(4)))  unsigned int u32x4;
typedef __attribute__((ext_vector_type(8)))  int i32x8;
typedef __attribute__((ext_vector_type(4)))  int i32x4;

#define NVOX 18432   // D*H*W = 8*48*48
#define DD 8
#define HH 48
#define WW 48
#define HW 2304      // HH*WW

#if defined(__has_builtin)
#  if __has_builtin(__builtin_amdgcn_tensor_load_to_lds) && \
      __has_builtin(__builtin_amdgcn_s_wait_tensorcnt)
#    define USE_TDM 1
#  endif
#endif

__device__ __forceinline__ unsigned short f2bf(float f) {
  unsigned u = __float_as_uint(f);
  u += 0x7FFFu + ((u >> 16) & 1u);          // round-to-nearest-even
  return (unsigned short)(u >> 16);
}

#ifdef USE_TDM
// Issue one TDM 2D tile load: bf16 tile [128 rows x 32 cols] from a packed
// weight matrix with row stride Kpad (elements) into LDS at ldsOff.
// D# fields per CDNA5 ISA 8.3/8.4: count=1, type=2(image), data_size=2B,
// tile_dim0=32, tile_dim1=128, tensor_dim1 = rows remaining (OOB rows -> 0).
__device__ __forceinline__ void tdm_load_A(const unsigned short* tileSrc,
                                           unsigned ldsOff, int Kpad,
                                           int dim0Remain, int rowsRemain) {
  unsigned long long ga = (unsigned long long)(const void*)tileSrc;
  u32x4 g0;
  g0[0] = 1u;                                            // count=1, user mode
  g0[1] = ldsOff;                                        // lds_addr (bytes)
  g0[2] = (unsigned)ga;                                  // global_addr[31:0]
  g0[3] = ((unsigned)(ga >> 32) & 0x01FFFFFFu) | (2u << 30); // addr[56:32]|type=2
  i32x8 g1;
  g1[0] = 0x00010000;                                    // data_size=1 (2 bytes)
  g1[1] = (int)(((unsigned)dim0Remain & 0xFFFFu) << 16); // tensor_dim0 lo16
  g1[2] = (int)((((unsigned)dim0Remain >> 16) & 0xFFFFu) |
                (((unsigned)rowsRemain & 0xFFFFu) << 16)); // dim0 hi | dim1 lo
  g1[3] = (int)((((unsigned)rowsRemain >> 16) & 0xFFFFu) |
                (32u << 16));                            // dim1 hi | tile_dim0=32
  g1[4] = 128;                                           // tile_dim1=128, tile_dim2=0
  g1[5] = Kpad;                                          // tensor_dim0_stride lo32
  g1[6] = 0;
  g1[7] = 0;
  i32x4 z4 = {0, 0, 0, 0};
#if __clang_major__ >= 23
  i32x8 z8 = {0, 0, 0, 0, 0, 0, 0, 0};
  __builtin_amdgcn_tensor_load_to_lds(g0, g1, z4, z4, z8, 0);
#else
  __builtin_amdgcn_tensor_load_to_lds(g0, g1, z4, z4, 0);
#endif
}
#endif

// ---------------------------------------------------------------------------
// Pack an fp32 weight matrix [M][srcStride] -> bf16 [Mpad][Kpad] (zero padded)
// ---------------------------------------------------------------------------
__global__ void pack_bf16_kernel(const float* __restrict__ src,
                                 unsigned short* __restrict__ dst,
                                 int M, int Mpad, int Ktake, int Kpad, int srcStride) {
  int total = Mpad * Kpad;
  for (int i = blockIdx.x * blockDim.x + threadIdx.x; i < total;
       i += gridDim.x * blockDim.x) {
    int m = i / Kpad, k = i - m * Kpad;
    float v = (m < M && k < Ktake) ? src[m * srcStride + k] : 0.0f;
    dst[i] = f2bf(v);
  }
}

// ---------------------------------------------------------------------------
// Global-pool branch folded into a per-output-channel bias for the wp GEMM.
// ---------------------------------------------------------------------------
__global__ void gpool_pbias_kernel(const float* __restrict__ x,
                                   const float* __restrict__ wg,
                                   const float* __restrict__ bg,
                                   const float* __restrict__ wp,
                                   const float* __restrict__ bp,
                                   float* __restrict__ pbias) {
  __shared__ float sbuf[256];
  __shared__ float g[16];
  __shared__ float brg[256];
  int t = threadIdx.x;
  int c = t >> 4, part = t & 15;
  float s = 0.f;
  for (int i = part; i < NVOX; i += 16) s += x[c * NVOX + i];
  sbuf[t] = s;
  __syncthreads();
  if (t < 16) {
    float a = 0.f;
    for (int j = 0; j < 16; ++j) a += sbuf[t * 16 + j];
    g[t] = a / (float)NVOX;
  }
  __syncthreads();
  float a = bg[t];
  for (int cc = 0; cc < 16; ++cc) a += wg[t * 16 + cc] * g[cc];
  brg[t] = fmaxf(a, 0.f);
  __syncthreads();
  float pb = bp[t];
  for (int cc = 0; cc < 256; ++cc) pb += wp[t * 1280 + 1024 + cc] * brg[cc];
  pbias[t] = pb;
}

// ---------------------------------------------------------------------------
// Implicit-im2col WMMA GEMM.  Block = 256 threads = 8 waves (4M x 2N), each
// wave owns a 32x32 output tile (2x2 WMMA frags) -> block tile 128M x 64N.
//   mode 0: dilated 3x3x3 from fp32 x (Ci=16)   mode 1: 1x1 from fp32 x
//   mode 2: 1x1 from bf16 concat                mode 3: 3x3x3 pad1 bf16 pyramid
//   epi 0: relu -> bf16        epi 1: offsets raw / sigmoid alpha split
// ---------------------------------------------------------------------------
__global__ __launch_bounds__(256) void wmma_gemm_kernel(
    const unsigned short* __restrict__ Apk, int Kpad, int Kreal, int Mpad,
    const float* __restrict__ bias, int Mvalid,
    const float* __restrict__ xsrc, const unsigned short* __restrict__ bsrc,
    int mode, int dil,
    unsigned short* __restrict__ outb, int rowOff,
    float* __restrict__ outOff, float* __restrict__ outAlpha, int epi) {
  __shared__ unsigned short As[128 * 32];   // [m][k] row-major, 8 KB
  __shared__ unsigned short Bs[64 * 32];    // [n][k] row-major, 4 KB
  const int tid  = threadIdx.x;
  const int lane = tid & 31;
  const int wave = tid >> 5;
  const int wm   = wave >> 1;        // 0..3
  const int wn   = wave & 1;         // 0..1
  const int half = (lane >> 4) & 1;
  const int l16  = lane & 15;
  const int nBase = blockIdx.x * 64;
  const int mBase = blockIdx.y * 128;

  // Per-thread voxel decomposition for B gather: this thread always gathers
  // k = tid&31 for the 8 columns n = (tid>>5) + 8j.  Registers only.
  const int kLoc  = tid & 31;
  const int nSlot = tid >> 5;
  int nzr[8], nyr[8], nxr[8];
  #pragma unroll
  for (int j = 0; j < 8; ++j) {
    int ng = nBase + nSlot + j * 8;
    int z = ng / HW;
    int rem = ng - z * HW;
    int y = rem / WW;
    nzr[j] = z; nyr[j] = y; nxr[j] = rem - y * WW;
  }

  v8f acc[2][2];
  #pragma unroll
  for (int mi = 0; mi < 2; ++mi)
    #pragma unroll
    for (int ni = 0; ni < 2; ++ni)
      acc[mi][ni] = (v8f){0.f, 0.f, 0.f, 0.f, 0.f, 0.f, 0.f, 0.f};

  for (int kt = 0; kt < Kpad; kt += 32) {
    // ---- stage A tile (128 x 32 bf16) ----
#ifdef USE_TDM
    if (wave == 0) {
      tdm_load_A(Apk + (size_t)mBase * Kpad + kt,
                 (unsigned)(size_t)(const void*)&As[0],
                 Kpad, Kpad - kt, Mpad - mBase);
    }
#else
    #pragma unroll
    for (int j = 0; j < 8; ++j) {
      int r  = (tid >> 4) + j * 16;
      int wd = tid & 15;
      unsigned val = 0;
      int gm = mBase + r;
      if (gm < Mpad) val = *(const unsigned*)&Apk[(size_t)gm * Kpad + kt + wd * 2];
      *(unsigned*)&As[r * 32 + wd * 2] = val;
    }
    if (kt + 32 < Kpad)
      __builtin_prefetch(&Apk[(size_t)(mBase + (tid & 127)) * Kpad + kt + 32], 0, 1);
#endif

    // ---- stage B tile (64 n x 32 k): one k-decomposition per thread ----
    int kk = kt + kLoc;
    bool kvalid = kk < Kreal;
    if (mode == 2) {
      const unsigned short* src = &bsrc[(size_t)kk * NVOX + nBase + nSlot];
      #pragma unroll
      for (int j = 0; j < 8; ++j)
        Bs[(nSlot + j * 8) * 32 + kLoc] = kvalid ? src[j * 8] : (unsigned short)0;
      if (kvalid && kt + 32 < Kreal)
        __builtin_prefetch(&bsrc[(size_t)(kk + 32) * NVOX + nBase + nSlot], 0, 1);
    } else if (mode == 1) {
      const float* src = &xsrc[(size_t)kk * NVOX + nBase + nSlot];
      #pragma unroll
      for (int j = 0; j < 8; ++j)
        Bs[(nSlot + j * 8) * 32 + kLoc] = kvalid ? f2bf(src[j * 8]) : (unsigned short)0;
    } else {                       // mode 0 (fp32 x, dilated) / 3 (bf16 pyr)
      int cch = kk / 27;
      int tap = kk - cch * 27;
      int ddz = (tap / 9 - 1) * dil;
      int ddy = ((tap / 3) % 3 - 1) * dil;
      int ddx = (tap % 3 - 1) * dil;
      size_t cb = (size_t)cch * NVOX;
      #pragma unroll
      for (int j = 0; j < 8; ++j) {
        int zz = nzr[j] + ddz, yy = nyr[j] + ddy, xx = nxr[j] + ddx;
        unsigned short hv = 0;
        if (kvalid && zz >= 0 && zz < DD && yy >= 0 && yy < HH && xx >= 0 && xx < WW) {
          size_t idx = cb + zz * HW + yy * WW + xx;
          hv = (mode == 0) ? f2bf(xsrc[idx]) : bsrc[idx];
        }
        Bs[(nSlot + j * 8) * 32 + kLoc] = hv;
      }
    }

#ifdef USE_TDM
    if (wave == 0) __builtin_amdgcn_s_wait_tensorcnt(0);
#endif
    __syncthreads();

    // ---- fragments per ISA 7.12.2 bf16 A/B layout, 2x2 WMMA ----
    union FR { v16bf v; unsigned u[8]; } a0, a1, b0, b1;
    const int aRow0 = wm * 32 + l16;
    const int bRow0 = wn * 32 + l16;
    #pragma unroll
    for (int v = 0; v < 8; ++v) {
      int kv = ((v < 4) ? 2 * v : 16 + 2 * (v - 4)) + half * 8;
      a0.u[v] = *(const unsigned*)&As[aRow0 * 32 + kv];
      a1.u[v] = *(const unsigned*)&As[(aRow0 + 16) * 32 + kv];
      b0.u[v] = *(const unsigned*)&Bs[bRow0 * 32 + kv];
      b1.u[v] = *(const unsigned*)&Bs[(bRow0 + 16) * 32 + kv];
    }
    acc[0][0] = __builtin_amdgcn_wmma_f32_16x16x32_bf16(
        false, a0.v, false, b0.v, (short)0, acc[0][0], false, false);
    acc[0][1] = __builtin_amdgcn_wmma_f32_16x16x32_bf16(
        false, a0.v, false, b1.v, (short)0, acc[0][1], false, false);
    acc[1][0] = __builtin_amdgcn_wmma_f32_16x16x32_bf16(
        false, a1.v, false, b0.v, (short)0, acc[1][0], false, false);
    acc[1][1] = __builtin_amdgcn_wmma_f32_16x16x32_bf16(
        false, a1.v, false, b1.v, (short)0, acc[1][1], false, false);
    __syncthreads();
  }

  // ---- epilogue: C layout vgpr r -> m = r + 8*half, n = lane%16 ----
  #pragma unroll
  for (int mi = 0; mi < 2; ++mi) {
    #pragma unroll
    for (int ni = 0; ni < 2; ++ni) {
      #pragma unroll
      for (int r = 0; r < 8; ++r) {
        int m = mBase + wm * 32 + mi * 16 + r + 8 * half;
        int n = nBase + wn * 32 + ni * 16 + l16;
        if (m >= Mvalid) continue;
        float v = acc[mi][ni][r] + bias[m];
        if (epi == 0) {
          outb[(size_t)(rowOff + m) * NVOX + n] = f2bf(fmaxf(v, 0.f));
        } else {
          if (m < 81) outOff[(size_t)m * NVOX + n] = v;
          else        outAlpha[(size_t)(m - 81) * NVOX + n] = 1.0f / (1.0f + __expf(-v));
        }
      }
    }
  }
}

// ---------------------------------------------------------------------------
// Modulated deformable conv3d: one thread per voxel, 27 taps, trilinear
// 8-corner sampling of 16 input channels, 16->32 dot with LDS-cached wdc.
// ---------------------------------------------------------------------------
__global__ __launch_bounds__(64) void deform_kernel(
    const float* __restrict__ x, const float* __restrict__ offw,
    const float* __restrict__ alph, const float* __restrict__ wdc,
    const float* __restrict__ bdc, float* __restrict__ out) {
  __shared__ float wds[32 * 16 * 27];
  for (int i = threadIdx.x; i < 32 * 16 * 27; i += 64) wds[i] = wdc[i];
  __syncthreads();
  int n = blockIdx.x * 64 + threadIdx.x;
  int z = n / HW;
  int rem = n - z * HW;
  int y = rem / WW;
  int xw = rem - y * WW;
  float acc[32];
  #pragma unroll
  for (int oc = 0; oc < 32; ++oc) acc[oc] = bdc[oc];

  #pragma unroll 1
  for (int tap = 0; tap < 27; ++tap) {
    int dz = tap / 9 - 1, dy = (tap / 3) % 3 - 1, dx = tap % 3 - 1;
    float pz = (float)(z + dz)  + offw[(size_t)(tap * 3 + 0) * NVOX + n];
    float py = (float)(y + dy)  + offw[(size_t)(tap * 3 + 1) * NVOX + n];
    float px = (float)(xw + dx) + offw[(size_t)(tap * 3 + 2) * NVOX + n];
    float a  = alph[(size_t)tap * NVOX + n];
    float z0f = floorf(pz), y0f = floorf(py), x0f = floorf(px);
    float fz = pz - z0f, fy = py - y0f, fx = px - x0f;
    int z0 = (int)z0f, y0 = (int)y0f, x0 = (int)x0f;
    float s[16];
    #pragma unroll
    for (int c = 0; c < 16; ++c) s[c] = 0.f;
    #pragma unroll
    for (int corner = 0; corner < 8; ++corner) {
      int cz = (corner >> 2) & 1, cy = (corner >> 1) & 1, cx = corner & 1;
      int zi = z0 + cz, yi = y0 + cy, xi = x0 + cx;
      if (zi < 0 || zi >= DD || yi < 0 || yi >= HH || xi < 0 || xi >= WW) continue;
      float wz = cz ? fz : 1.f - fz;
      float wy = cy ? fy : 1.f - fy;
      float wx = cx ? fx : 1.f - fx;
      float w = wz * wy * wx;
      int idx = zi * HW + yi * WW + xi;
      #pragma unroll
      for (int c = 0; c < 16; ++c) s[c] += x[(size_t)c * NVOX + idx] * w;
    }
    #pragma unroll 1
    for (int oc = 0; oc < 32; ++oc) {
      float t = 0.f;
      #pragma unroll
      for (int c = 0; c < 16; ++c) t += s[c] * wds[(oc * 16 + c) * 27 + tap];
      acc[oc] += t * a;
    }
  }
  #pragma unroll
  for (int oc = 0; oc < 32; ++oc) out[(size_t)oc * NVOX + n] = acc[oc];
}

// ---------------------------------------------------------------------------
extern "C" void kernel_launch(void* const* d_in, const int* in_sizes, int n_in,
                              void* d_out, int out_size, void* d_ws, size_t ws_size,
                              hipStream_t stream) {
  (void)in_sizes; (void)n_in; (void)out_size;
  const float* x    = (const float*)d_in[0];
  const float* w1   = (const float*)d_in[1];
  const float* b1   = (const float*)d_in[2];
  const float* w2   = (const float*)d_in[3];
  const float* b2   = (const float*)d_in[4];
  const float* w3   = (const float*)d_in[5];
  const float* b3   = (const float*)d_in[6];
  const float* w4   = (const float*)d_in[7];
  const float* b4   = (const float*)d_in[8];
  const float* wg   = (const float*)d_in[9];
  const float* bg   = (const float*)d_in[10];
  const float* wp   = (const float*)d_in[11];
  const float* bp   = (const float*)d_in[12];
  const float* wdef = (const float*)d_in[13];
  const float* bdef = (const float*)d_in[14];
  const float* wdc  = (const float*)d_in[15];
  const float* bdc  = (const float*)d_in[16];
  float* out = (float*)d_out;

  char* ws = (char*)d_ws;
  size_t off = 0;
  auto alloc = [&](size_t bytes) -> char* {
    char* p = ws + off;
    off = (off + bytes + 255) & ~(size_t)255;
    return p;
  };
  unsigned short* CAT  = (unsigned short*)alloc((size_t)1024 * NVOX * 2);
  unsigned short* PYR  = (unsigned short*)alloc((size_t)256  * NVOX * 2);
  float*          OFFW = (float*)         alloc((size_t)81   * NVOX * 4);
  float*          ALPW = (float*)         alloc((size_t)27   * NVOX * 4);
  float*          PBIA = (float*)         alloc(256 * 4);
  unsigned short* W1P  = (unsigned short*)alloc((size_t)256 * 32   * 2);
  unsigned short* W2P  = (unsigned short*)alloc((size_t)256 * 448  * 2);
  unsigned short* W3P  = (unsigned short*)alloc((size_t)256 * 448  * 2);
  unsigned short* W4P  = (unsigned short*)alloc((size_t)256 * 448  * 2);
  unsigned short* WPP  = (unsigned short*)alloc((size_t)256 * 1024 * 2);
  unsigned short* WDP  = (unsigned short*)alloc((size_t)112 * 6912 * 2);
  if (off > ws_size) return;   // scratch too small: bail deterministically

  // weight packing
  pack_bf16_kernel<<<64,  256, 0, stream>>>(w1,   W1P, 256, 256, 16,   32,   16);
  pack_bf16_kernel<<<256, 256, 0, stream>>>(w2,   W2P, 256, 256, 432,  448,  432);
  pack_bf16_kernel<<<256, 256, 0, stream>>>(w3,   W3P, 256, 256, 432,  448,  432);
  pack_bf16_kernel<<<256, 256, 0, stream>>>(w4,   W4P, 256, 256, 432,  448,  432);
  pack_bf16_kernel<<<512, 256, 0, stream>>>(wp,   WPP, 256, 256, 1024, 1024, 1280);
  pack_bf16_kernel<<<512, 256, 0, stream>>>(wdef, WDP, 108, 112, 6912, 6912, 6912);

  // global-pool branch folded into wp bias
  gpool_pbias_kernel<<<1, 256, 0, stream>>>(x, wg, bg, wp, bp, PBIA);

  dim3 gA(NVOX / 64, 2);   // 288 x 2 : block tile 128M x 64N, M=256
  // ASPP branches -> concat rows [0,1024) in bf16
  wmma_gemm_kernel<<<gA, 256, 0, stream>>>(W1P, 32,   16,  256, b1, 256, x, nullptr, 1, 0,
                                           CAT, 0,   nullptr, nullptr, 0);
  wmma_gemm_kernel<<<gA, 256, 0, stream>>>(W2P, 448,  432, 256, b2, 256, x, nullptr, 0, 6,
                                           CAT, 256, nullptr, nullptr, 0);
  wmma_gemm_kernel<<<gA, 256, 0, stream>>>(W3P, 448,  432, 256, b3, 256, x, nullptr, 0, 12,
                                           CAT, 512, nullptr, nullptr, 0);
  wmma_gemm_kernel<<<gA, 256, 0, stream>>>(W4P, 448,  432, 256, b4, 256, x, nullptr, 0, 18,
                                           CAT, 768, nullptr, nullptr, 0);
  // projection 1x1: concat(1024) -> pyramid(256), bias = pbias (pool branch folded)
  wmma_gemm_kernel<<<gA, 256, 0, stream>>>(WPP, 1024, 1024, 256, PBIA, 256, nullptr, CAT, 2, 0,
                                           PYR, 0,   nullptr, nullptr, 0);
  // defo conv 3x3x3 pad1: pyramid(256ch) -> 108 ch (81 offsets raw, 27 sigmoid alpha)
  dim3 gD(NVOX / 64, 1);   // Mpad=112 fits one 128-row block tile
  wmma_gemm_kernel<<<gD, 256, 0, stream>>>(WDP, 6912, 6912, 112, bdef, 108, nullptr, PYR, 3, 1,
                                           nullptr, 0, OFFW, ALPW, 1);
  // final modulated deformable conv
  deform_kernel<<<NVOX / 64, 64, 0, stream>>>(x, OFFW, ALPW, wdc, bdc, out);
}
